// CausalSelfAttention_47974784696991
// MI455X (gfx1250) — matrix operand order repaired and checked by reading, not compiled, not run on hardware
//
#include <hip/hip_runtime.h>

typedef __attribute__((ext_vector_type(16))) _Float16 v16h;
typedef __attribute__((ext_vector_type(8)))  float    v8f;
typedef __attribute__((ext_vector_type(4)))  int      i4v;

struct alignas(16) U4 { unsigned a, b, c, d; };
union FragH { v16h v; U4 u[2]; };
union H8   { U4 u; _Float16 h[8]; };

// ---- async global->LDS copy (gfx1250 GLOBAL_LOAD_ASYNC_TO_LDS_B128, ASYNCcnt) ----
#if defined(__has_builtin)
#if __has_builtin(__builtin_amdgcn_global_load_async_to_lds_b128) && \
    __has_builtin(__builtin_amdgcn_s_wait_asynccnt)
#define USE_ASYNC_COPY 1
#endif
#endif
#ifndef USE_ASYNC_COPY
#define USE_ASYNC_COPY 0
#endif

#if USE_ASYNC_COPY
typedef __attribute__((address_space(1))) i4v GI4;   // global int4*
typedef __attribute__((address_space(3))) i4v LI4;   // LDS int4*
__device__ __forceinline__ void copy16_g2l(const void* g, void* l) {
  __builtin_amdgcn_global_load_async_to_lds_b128((GI4*)g, (LI4*)l, 0, 0);
}
__device__ __forceinline__ void copy_wait() { __builtin_amdgcn_s_wait_asynccnt(0); }
#else
__device__ __forceinline__ void copy16_g2l(const void* g, void* l) {
  *(U4*)l = *(const U4*)g;
}
__device__ __forceinline__ void copy_wait() {}
#endif

__device__ __forceinline__ unsigned pack2(float x, float y) {
  union { _Float16 h[2]; unsigned u; } p;
  p.h[0] = (_Float16)x; p.h[1] = (_Float16)y;
  return p.u;
}
__device__ __forceinline__ U4 pack8(float4 f0, float4 f1) {
  U4 q;
  q.a = pack2(f0.x, f0.y); q.b = pack2(f0.z, f0.w);
  q.c = pack2(f1.x, f1.y); q.d = pack2(f1.z, f1.w);
  return q;
}

// A-matrix fragment (16x32 f16, MxK), tile stored row-major [16][stride] at (row0,k0).
// ISA: lane<16 holds row=lane, K in {0..7,16..23}; lane>=16 same rows, K shifted +8.
__device__ __forceinline__ v16h frag_a16x32(const _Float16* tile, int stride, int lane) {
  const int row = lane & 15;
  const int hi  = lane >> 4;
  const _Float16* p = tile + row * stride + hi * 8;
  FragH f;
  f.u[0] = *(const U4*)(p);
  f.u[1] = *(const U4*)(p + 16);
  return f.v;
}

// B-matrix fragment (32x16 f16, KxN), tile stored TRANSPOSED [N][K] row-major at (n0,k0).
// ISA: lane<16 holds col N=lane&15, K in {0..7,16..23}; lane>=16 same cols, K shifted +8.
__device__ __forceinline__ v16h frag_b32x16(const _Float16* tileT, int stride, int lane) {
  const int col = lane & 15;
  const int hi  = lane >> 4;
  const _Float16* p = tileT + col * stride + hi * 8;
  FragH f;
  f.u[0] = *(const U4*)(p);
  f.u[1] = *(const U4*)(p + 16);
  return f.v;
}

__device__ __forceinline__ v8f wmma_f16(v16h a, v16h b, v8f c) {
  return __builtin_amdgcn_wmma_f32_16x16x32_f16(false, a, false, b, (short)0, c, false, false);
}

// ---------------------------------------------------------------------------
// GEMM: C[M,N] = A[M,K] @ W[K,N] + bias[N].  A: f32 or f16; W,bias: f32;
// Cout: f32 or f16.  f16 WMMA with f32 accumulate.
// BM=128, BN=64, BK=64; 256 threads = 8 waves (4Mx2N); wave tile 32x32.
// ---------------------------------------------------------------------------
template <typename TA, typename TOUT>
__global__ __launch_bounds__(256)
void gemm_bias_wmma(const TA* __restrict__ A, const float* __restrict__ W,
                    const float* __restrict__ bias, TOUT* __restrict__ Cout,
                    int M, int N, int K) {
  constexpr int BM = 128, BN = 64, BK = 64;
  constexpr int AS = 72, BS = 72;            // strides in halves (144B, 16B-aligned)
  __shared__ _Float16 As[BM * AS];            // [m][k]
  __shared__ _Float16 Bt[BN * BS];            // [n][k] (transposed)

  const int tid  = threadIdx.x;
  const int lane = tid & 31;
  const int wave = tid >> 5;
  const int wm = wave >> 1;                   // 0..3
  const int wn = wave & 1;                    // 0..1
  const int m0 = blockIdx.x * BM;
  const int n0 = blockIdx.y * BN;

  const v8f vz = {0.f,0.f,0.f,0.f,0.f,0.f,0.f,0.f};
  v8f acc[2][2];
  acc[0][0] = vz; acc[0][1] = vz; acc[1][0] = vz; acc[1][1] = vz;

  const int arow = tid >> 1;                  // 0..127
  const int acol = (tid & 1) * 32;            // 0 / 32   (32 halves per thread)
  const int brow = tid >> 2;                  // k: 0..63
  const int bcol = (tid & 3) * 16;            // n: 0,16,32,48

  const TA*    gpA = A + (size_t)(m0 + arow) * K + acol;
  const float* gpB = W + (size_t)brow * N + n0 + bcol;

  for (int k0 = 0; k0 < K; k0 += BK) {
    {   // stage A tile (row-major f16)
      _Float16* sp = &As[arow * AS + acol];
      if constexpr (sizeof(TA) == 4) {        // convert f32 -> f16, packed b128 stores
        const float* g = (const float*)gpA;
        #pragma unroll
        for (int i = 0; i < 4; ++i) {
          float4 f0 = *(const float4*)(g + i * 8);
          float4 f1 = *(const float4*)(g + i * 8 + 4);
          *(U4*)(sp + i * 8) = pack8(f0, f1);
        }
      } else {                                // pure copy: async global->LDS
        #pragma unroll
        for (int i = 0; i < 4; ++i) copy16_g2l(gpA + i * 8, sp + i * 8);
      }
    }
    {   // stage B tile transposed: Bt[n][k] = W[k0+k][n0+n]
      #pragma unroll
      for (int i = 0; i < 4; ++i) {
        float4 f = *(const float4*)(gpB + i * 4);
        Bt[(bcol + i*4 + 0) * BS + brow] = (_Float16)f.x;
        Bt[(bcol + i*4 + 1) * BS + brow] = (_Float16)f.y;
        Bt[(bcol + i*4 + 2) * BS + brow] = (_Float16)f.z;
        Bt[(bcol + i*4 + 3) * BS + brow] = (_Float16)f.w;
      }
    }
    gpA += BK;
    gpB += (size_t)BK * N;
    if (k0 + BK < K) {                        // prefetch next tiles
      __builtin_prefetch(gpA, 0, 1);
      __builtin_prefetch(gpB, 0, 1);
    }
    if constexpr (sizeof(TA) == 2) copy_wait();
    __syncthreads();

    #pragma unroll
    for (int ks = 0; ks < BK; ks += 32) {
      v16h a0 = frag_a16x32(&As[(wm*32 +  0) * AS + ks], AS, lane);
      v16h a1 = frag_a16x32(&As[(wm*32 + 16) * AS + ks], AS, lane);
      v16h b0 = frag_b32x16(&Bt[(wn*32 +  0) * BS + ks], BS, lane);
      v16h b1 = frag_b32x16(&Bt[(wn*32 + 16) * BS + ks], BS, lane);
      acc[0][0] = wmma_f16(a0, b0, acc[0][0]);
      acc[0][1] = wmma_f16(a0, b1, acc[0][1]);
      acc[1][0] = wmma_f16(a1, b0, acc[1][0]);
      acc[1][1] = wmma_f16(a1, b1, acc[1][1]);
    }
    __syncthreads();
  }

  // epilogue: C layout row = r + 8*(lane>>4), col = lane&15
  const int hi = lane >> 4;
  #pragma unroll
  for (int mi = 0; mi < 2; ++mi) {
    #pragma unroll
    for (int ni = 0; ni < 2; ++ni) {
      const int col = n0 + wn*32 + ni*16 + (lane & 15);
      const float bval = bias[col];
      #pragma unroll
      for (int r = 0; r < 8; ++r) {
        const int row = m0 + wm*32 + mi*16 + r + hi*8;
        Cout[(size_t)row * N + col] = (TOUT)(acc[mi][ni][r] + bval);
      }
    }
  }
}

// ---------------------------------------------------------------------------
// Flash attention over f16 qkv [B][T][3][H][Dh].  Per (b, h, 128-query tile);
// 8 waves x 16 query rows; 64-key blocks, causal, online softmax.
// Writes f16 y[b][t][h*Dh + d]  ([B,T,C], feeds the output projection).
// ---------------------------------------------------------------------------
__global__ __launch_bounds__(256)
void flash_attn_f16wmma(const _Float16* __restrict__ qkv, _Float16* __restrict__ y) {
  constexpr int T = 2048, C = 1024, Dh = 64;
  constexpr int BQ = 128, BKEY = 64, ST = 72;   // 144B stride, 16B-aligned
  __shared__ _Float16 Qs[BQ * ST];              // [qrow][d]      (18 KB)
  __shared__ _Float16 Ks[BKEY * ST];            // [key][d]       ( 9 KB)
  __shared__ _Float16 Vt[Dh * ST];              // [d][key]       ( 9 KB)
  __shared__ _Float16 Ps[8 * 16 * ST];          // per-wave [row][key] (18 KB)

  const int tid  = threadIdx.x;
  const int lane = tid & 31;
  const int wave = tid >> 5;
  const int hi   = lane >> 4;
  const int lcol = lane & 15;
  const int b  = blockIdx.z;
  const int h  = blockIdx.y;
  const int qbase = blockIdx.x * BQ;

  const size_t rowstride = (size_t)3 * C;       // halves per timestep
  const _Float16* base = qkv + (size_t)b * T * rowstride + h * Dh;
  // q at +0, k at +C, v at +2C within a timestep row

  {   // stage Q tile: pure 16B copies -> async to LDS (scale applied to scores)
    const int row = tid >> 1;                   // 0..127
    const int dof = (tid & 1) * 32;
    const _Float16* gp = base + (size_t)(qbase + row) * rowstride + dof;
    _Float16* sp = &Qs[row * ST + dof];
    #pragma unroll
    for (int i = 0; i < 4; ++i) copy16_g2l(gp + i * 8, sp + i * 8);
  }

  const v8f vz = {0.f,0.f,0.f,0.f,0.f,0.f,0.f,0.f};
  float mrow[8], lrow[8];
  v8f o[4];
  #pragma unroll
  for (int r = 0; r < 8; ++r) { mrow[r] = -3.0e38f; lrow[r] = 0.f; }
  #pragma unroll
  for (int t = 0; t < 4; ++t) o[t] = vz;

  const int m0 = wave * 16;
  const int kblocks = (qbase + BQ) / BKEY;      // uniform per block (causal bound)

  for (int jb = 0; jb < kblocks; ++jb) {
    const int kbase = jb * BKEY;
    __syncthreads();                            // K/V region free for rewrite
    {   // stage K [key][d] (async copy) and V transposed [d][key] (via regs)
      const int row = tid >> 2;                 // key 0..63
      const int dof = (tid & 3) * 16;
      const _Float16* gk = base + (size_t)(kbase + row) * rowstride + C   + dof;
      const U4*       gv = (const U4*)(base + (size_t)(kbase + row) * rowstride + 2*C + dof);
      _Float16* sk = &Ks[row * ST + dof];
      copy16_g2l(gk,     sk);
      copy16_g2l(gk + 8, sk + 8);
      #pragma unroll
      for (int i = 0; i < 2; ++i) {
        H8 v; v.u = gv[i];
        #pragma unroll
        for (int j = 0; j < 8; ++j)
          Vt[(dof + i*8 + j) * ST + row] = v.h[j];
      }
    }
    copy_wait();                                // ASYNCcnt -> 0 (covers Q on iter 0)
    __syncthreads();

    // S = Q(16x64) @ K^T(64x64): N = key, K-dim = d (Ks rows are keys -> contiguous d)
    v16h aq0 = frag_a16x32(&Qs[m0 * ST +  0], ST, lane);
    v16h aq1 = frag_a16x32(&Qs[m0 * ST + 32], ST, lane);
    v8f s[4];
    #pragma unroll
    for (int t = 0; t < 4; ++t) {
      v16h bk0 = frag_b32x16(&Ks[(t*16) * ST +  0], ST, lane);
      v16h bk1 = frag_b32x16(&Ks[(t*16) * ST + 32], ST, lane);
      s[t] = wmma_f16(aq0, bk0, vz);
      s[t] = wmma_f16(aq1, bk1, s[t]);
    }

    // scale (1/sqrt(Dh)) + causal mask
    #pragma unroll
    for (int t = 0; t < 4; ++t) {
      const int key = kbase + t*16 + lcol;
      #pragma unroll
      for (int r = 0; r < 8; ++r) {
        const int qrow = qbase + m0 + r + hi*8;
        s[t][r] = (key > qrow) ? -3.0e38f : s[t][r] * 0.125f;
      }
    }

    // online softmax (rows live in 16-lane halves; reduce with xor shuffles)
    #pragma unroll
    for (int r = 0; r < 8; ++r) {
      float mx = fmaxf(fmaxf(s[0][r], s[1][r]), fmaxf(s[2][r], s[3][r]));
      mx = fmaxf(mx, __shfl_xor(mx, 1));
      mx = fmaxf(mx, __shfl_xor(mx, 2));
      mx = fmaxf(mx, __shfl_xor(mx, 4));
      mx = fmaxf(mx, __shfl_xor(mx, 8));
      const float mnew  = fmaxf(mrow[r], mx);
      const float alpha = __expf(mrow[r] - mnew);
      mrow[r] = mnew;
      float psum = 0.f;
      #pragma unroll
      for (int t = 0; t < 4; ++t) {
        const float p = __expf(s[t][r] - mnew);
        s[t][r] = p;
        psum += p;
      }
      psum += __shfl_xor(psum, 1);
      psum += __shfl_xor(psum, 2);
      psum += __shfl_xor(psum, 4);
      psum += __shfl_xor(psum, 8);
      lrow[r] = lrow[r] * alpha + psum;
      #pragma unroll
      for (int t = 0; t < 4; ++t) o[t][r] *= alpha;
    }

    // re-layout P (C-layout regs -> A-fragment layout) through per-wave LDS
    #pragma unroll
    for (int t = 0; t < 4; ++t)
      #pragma unroll
      for (int r = 0; r < 8; ++r)
        Ps[(wave*16 + r + hi*8) * ST + t*16 + lcol] = (_Float16)s[t][r];
    __syncthreads();

    // O += P(16x64) @ V(64x64): N = d, K-dim = key (Vt rows are d -> contiguous key)
    v16h ap0 = frag_a16x32(&Ps[(wave*16) * ST +  0], ST, lane);
    v16h ap1 = frag_a16x32(&Ps[(wave*16) * ST + 32], ST, lane);
    #pragma unroll
    for (int t = 0; t < 4; ++t) {
      v16h bv0 = frag_b32x16(&Vt[(t*16) * ST +  0], ST, lane);
      v16h bv1 = frag_b32x16(&Vt[(t*16) * ST + 32], ST, lane);
      o[t] = wmma_f16(ap0, bv0, o[t]);
      o[t] = wmma_f16(ap1, bv1, o[t]);
    }
  }

  // final: y[b][qrow][h*Dh + d] = O / l   (f16)
  #pragma unroll
  for (int t = 0; t < 4; ++t) {
    const int d = t*16 + lcol;
    #pragma unroll
    for (int r = 0; r < 8; ++r) {
      const int qrow = qbase + m0 + r + hi*8;
      y[((size_t)b * T + qrow) * C + h * Dh + d] = (_Float16)(o[t][r] / lrow[r]);
    }
  }
}

// ---------------------------------------------------------------------------
extern "C" void kernel_launch(void* const* d_in, const int* in_sizes, int n_in,
                              void* d_out, int out_size, void* d_ws, size_t ws_size,
                              hipStream_t stream) {
  (void)in_sizes; (void)n_in; (void)out_size; (void)ws_size;
  constexpr int B = 4, T = 2048, C = 1024;
  constexpr int M = B * T;                      // 8192

  const float* x     = (const float*)d_in[0];
  const float* w_qkv = (const float*)d_in[1];
  const float* b_qkv = (const float*)d_in[2];
  const float* w_out = (const float*)d_in[3];
  const float* b_out = (const float*)d_in[4];
  float* out = (float*)d_out;

  _Float16* qkv = (_Float16*)d_ws;              // [B,T,3C] f16  48 MB
  _Float16* yb  = qkv + (size_t)M * 3 * C;      // [B,T,C]  f16  16 MB

  // 1) qkv = f16(x @ w_qkv + b_qkv)       (8192 x 3072 x 1024)
  gemm_bias_wmma<float, _Float16><<<dim3(M / 128, (3 * C) / 64), 256, 0, stream>>>(
      x, w_qkv, b_qkv, qkv, M, 3 * C, C);

  // 2) flash attention                    (grid: 16 q-tiles x 16 heads x 4 batch)
  flash_attn_f16wmma<<<dim3(T / 128, 16, B), 256, 0, stream>>>(qkv, yb);

  // 3) out = y @ w_out + b_out            (8192 x 1024 x 1024, f32 out)
  gemm_bias_wmma<_Float16, float><<<dim3(M / 128, C / 64), 256, 0, stream>>>(
      yb, w_out, b_out, out, M, C, C);
}
